// density_specified_OT_loss_37065567765210
// MI455X (gfx1250) — compile-verified, matrix operand order checked by reading, not run on hardware
//
#include <hip/hip_runtime.h>
#include <hip/hip_bf16.h>

typedef __attribute__((ext_vector_type(16))) _Float16 v16h;
typedef __attribute__((ext_vector_type(8)))  _Float16 v8h;
typedef __attribute__((ext_vector_type(8)))  float    v8f;

#define NPTS 8192
#define DIM  32
#define INV_REG 10.0f     // 1/0.1
#define NITER 100
#define JSEG 8            // j-segments per matvec pass (parallelism)
#define VSCALE 2048.0f    // fp16 carrier scale for u,v (centers fp16 dynamic range)

// ---------- prep: f32 -> fp16 rows + squared norms (wave per row) ----------
__global__ void cvt_rows(const float* __restrict__ x, _Float16* __restrict__ xh,
                         float* __restrict__ x2) {
    int lane = threadIdx.x & 31;
    int wave = threadIdx.x >> 5;
    int row  = blockIdx.x * 8 + wave;
    float v = x[(size_t)row * DIM + lane];
    xh[(size_t)row * DIM + lane] = (_Float16)v;
    float s = v * v;
    #pragma unroll
    for (int off = 16; off > 0; off >>= 1) s += __shfl_down(s, off, 32);
    if (lane == 0) x2[row] = s;
}

// ---------- softmax over 8192 elements, single block ----------
__global__ void softmax8192(const float* __restrict__ in, float* __restrict__ out) {
    __shared__ float red[256];
    int t = threadIdx.x;
    float mx = -3.4e38f;
    for (int i = t; i < NPTS; i += 256) mx = fmaxf(mx, in[i]);
    red[t] = mx; __syncthreads();
    for (int s = 128; s > 0; s >>= 1) { if (t < s) red[t] = fmaxf(red[t], red[t + s]); __syncthreads(); }
    mx = red[0]; __syncthreads();
    float sum = 0.f;
    for (int i = t; i < NPTS; i += 256) sum += __expf(in[i] - mx);
    red[t] = sum; __syncthreads();
    for (int s = 128; s > 0; s >>= 1) { if (t < s) red[t] += red[t + s]; __syncthreads(); }
    float inv = 1.0f / red[0];
    for (int i = t; i < NPTS; i += 256) out[i] = __expf(in[i] - mx) * inv;
}

// ---------- init: u = 1/N (f32), u_hat = VSCALE*u (f16), zero loss ----------
__global__ void init_uv(float* __restrict__ uf, _Float16* __restrict__ uh,
                        float* __restrict__ out) {
    int i = blockIdx.x * 256 + threadIdx.x;
    uf[i] = 1.0f / (float)NPTS;
    uh[i] = (_Float16)(VSCALE / (float)NPTS);
    if (i == 0) out[0] = 0.f;
}

// ---------- build K and K^T (fp16) via WMMA: K = exp(-10*sqrt(x2+y2-2 x.y)) ----------
__global__ void build_K(const _Float16* __restrict__ xh, const _Float16* __restrict__ yh,
                        const float* __restrict__ x2, const float* __restrict__ y2,
                        _Float16* __restrict__ K, _Float16* __restrict__ KT) {
    __shared__ _Float16 tile[16][136];   // 16 i-rows x 128 j-cols (+pad), 272B pitch (16B aligned)
    int lane = threadIdx.x & 31;
    int wave = threadIdx.x >> 5;
    int i0 = blockIdx.y * 16;
    int j0 = blockIdx.x * 128;
    int jw = j0 + wave * 16;
    int hi = lane >> 4;     // selects K-halves per ISA 16-bit 16x32 layout
    int lo = lane & 15;

    union Frag { v16h v; v8h h[2]; };
    const v8h* ap = (const v8h*)(xh + (size_t)(i0 + lo) * DIM + hi * 8);
    Frag fa; fa.h[0] = ap[0]; fa.h[1] = ap[2];            // k = hi*8..+7 and 16+hi*8..+7
    const v8h* bp = (const v8h*)(yh + (size_t)(jw + lo) * DIM + hi * 8);
    Frag fb; fb.h[0] = bp[0]; fb.h[1] = bp[2];

    v8f c = {};
    c = __builtin_amdgcn_wmma_f32_16x16x32_f16(false, fa.v, false, fb.v,
                                               (short)0, c, false, false);

    float y2v = y2[jw + lo];
    #pragma unroll
    for (int r = 0; r < 8; ++r) {
        int ir = r + hi * 8;
        float d2 = x2[i0 + ir] + y2v - 2.0f * c[r];
        float d = __builtin_amdgcn_sqrtf(fmaxf(d2, 1e-12f));
        tile[ir][wave * 16 + lo] = (_Float16)__expf(-INV_REG * d);
    }
    __syncthreads();

    // K: coalesced 16B stores, thread t -> row t>>4, seg t&15
    {
        int row = threadIdx.x >> 4, seg = threadIdx.x & 15;
        v8h val = *(const v8h*)&tile[row][seg * 8];
        *(v8h*)(K + (size_t)(i0 + row) * NPTS + j0 + seg * 8) = val;
    }
    // K^T: thread t -> transposed row (j), 8 i-values gathered from LDS
    {
        int jrow = threadIdx.x >> 1, half = threadIdx.x & 1;
        v8h val;
        #pragma unroll
        for (int e = 0; e < 8; ++e) val[e] = tile[half * 8 + e][jrow];
        *(v8h*)(KT + (size_t)(j0 + jrow) * NPTS + i0 + half * 8) = val;
    }
}

// ---------- WMMA matvec partial: partial[seg][row] = sum_{j in seg} Mtx[row][j]*in_h[j] ----
// A = Mtx tile (16x32), B = in_h broadcast into all 16 columns -> every C column = row sum.
__global__ void pass_partial(const _Float16* __restrict__ Mtx,
                             const _Float16* __restrict__ in_h,
                             float* __restrict__ partial) {
    int lane = threadIdx.x & 31;
    int wave = threadIdx.x >> 5;
    int hi = lane >> 4;
    int lo = lane & 15;
    int rowbase = blockIdx.x * 128 + wave * 16;
    int jseg = blockIdx.y;
    int jbeg = jseg * (NPTS / JSEG);          // 1024-wide segment -> 32 k-steps

    const _Float16* Kp = Mtx + (size_t)(rowbase + lo) * NPTS + hi * 8;
    const _Float16* Vp = in_h + hi * 8;

    union Frag { v16h v; v8h h[2]; };
    v8f c0 = {}, c1 = {};
    #pragma unroll 2
    for (int j = jbeg; j < jbeg + NPTS / JSEG; j += 64) {
        Frag a0, b0, a1, b1;
        a0.h[0] = *(const v8h*)(Kp + j);       a0.h[1] = *(const v8h*)(Kp + j + 16);
        b0.h[0] = *(const v8h*)(Vp + j);       b0.h[1] = *(const v8h*)(Vp + j + 16);
        a1.h[0] = *(const v8h*)(Kp + j + 32);  a1.h[1] = *(const v8h*)(Kp + j + 48);
        b1.h[0] = *(const v8h*)(Vp + j + 32);  b1.h[1] = *(const v8h*)(Vp + j + 48);
        c0 = __builtin_amdgcn_wmma_f32_16x16x32_f16(false, a0.v, false, b0.v,
                                                    (short)0, c0, false, false);
        c1 = __builtin_amdgcn_wmma_f32_16x16x32_f16(false, a1.v, false, b1.v,
                                                    (short)0, c1, false, false);
    }
    if (lo == 0) {   // column 0 holds the row sums (all columns identical)
        #pragma unroll
        for (int r = 0; r < 8; ++r) {
            int row = rowbase + r + 8 * hi;
            partial[(size_t)jseg * NPTS + row] = c0[r] + c1[r];
        }
    }
}

// ---------- finish: s carries VSCALE from in_h; out_f = VSCALE*w/s (true value),
// ---------- out_h = VSCALE*out_f (scaled fp16 carrier for the next pass) ----------
__global__ void pass_finish(const float* __restrict__ partial, const float* __restrict__ w,
                            float* __restrict__ out_f, _Float16* __restrict__ out_h) {
    int row = blockIdx.x * 256 + threadIdx.x;
    float s = 0.f;
    #pragma unroll
    for (int seg = 0; seg < JSEG; ++seg) s += partial[(size_t)seg * NPTS + row];
    float of = VSCALE * w[row] / s;
    out_f[row] = of;
    out_h[row] = (_Float16)(of * VSCALE);
}

// ---------- loss: recompute d exactly, loss += sum u_i * exp(-10d) * d * v_j ----------
__global__ void loss_kernel(const _Float16* __restrict__ xh, const _Float16* __restrict__ yh,
                            const float* __restrict__ x2, const float* __restrict__ y2,
                            const float* __restrict__ u, const float* __restrict__ v,
                            float* __restrict__ out) {
    __shared__ float wsum[8];
    int lane = threadIdx.x & 31;
    int wave = threadIdx.x >> 5;
    int i0 = blockIdx.y * 16;
    int jw = blockIdx.x * 128 + wave * 16;
    int hi = lane >> 4;
    int lo = lane & 15;

    union Frag { v16h v; v8h h[2]; };
    const v8h* ap = (const v8h*)(xh + (size_t)(i0 + lo) * DIM + hi * 8);
    Frag fa; fa.h[0] = ap[0]; fa.h[1] = ap[2];
    const v8h* bp = (const v8h*)(yh + (size_t)(jw + lo) * DIM + hi * 8);
    Frag fb; fb.h[0] = bp[0]; fb.h[1] = bp[2];

    v8f c = {};
    c = __builtin_amdgcn_wmma_f32_16x16x32_f16(false, fa.v, false, fb.v,
                                               (short)0, c, false, false);

    float y2v = y2[jw + lo];
    float vj  = v[jw + lo];
    float acc = 0.f;
    #pragma unroll
    for (int r = 0; r < 8; ++r) {
        int i = i0 + r + 8 * hi;
        float d2 = x2[i] + y2v - 2.0f * c[r];
        float d = __builtin_amdgcn_sqrtf(fmaxf(d2, 1e-12f));
        float k = __expf(-INV_REG * d);
        acc += u[i] * k * d * vj;
    }
    #pragma unroll
    for (int off = 16; off > 0; off >>= 1) acc += __shfl_down(acc, off, 32);
    if (lane == 0) wsum[wave] = acc;
    __syncthreads();
    if (threadIdx.x == 0) {
        float s = 0.f;
        #pragma unroll
        for (int wv = 0; wv < 8; ++wv) s += wsum[wv];
        atomicAdd(out, s);
    }
}

extern "C" void kernel_launch(void* const* d_in, const int* in_sizes, int n_in,
                              void* d_out, int out_size, void* d_ws, size_t ws_size,
                              hipStream_t stream) {
    const float* src   = (const float*)d_in[0];   // [8192,32]
    const float* tgt   = (const float*)d_in[1];   // [8192,32]
    const float* sdens = (const float*)d_in[2];   // [8192]
    const float* tdens = (const float*)d_in[3];   // [8192]
    float* out = (float*)d_out;

    char* ws = (char*)d_ws;
    size_t off = 0;
    auto carve = [&](size_t bytes) -> char* {
        char* p = ws + off;
        off += (bytes + 255) & ~(size_t)255;
        return p;
    };
    _Float16* K  = (_Float16*)carve((size_t)NPTS * NPTS * sizeof(_Float16)); // 128 MB
    _Float16* KT = (_Float16*)carve((size_t)NPTS * NPTS * sizeof(_Float16)); // 128 MB
    _Float16* xh = (_Float16*)carve((size_t)NPTS * DIM * sizeof(_Float16));
    _Float16* yh = (_Float16*)carve((size_t)NPTS * DIM * sizeof(_Float16));
    float* x2 = (float*)carve(NPTS * sizeof(float));
    float* y2 = (float*)carve(NPTS * sizeof(float));
    float* mu = (float*)carve(NPTS * sizeof(float));
    float* nu = (float*)carve(NPTS * sizeof(float));
    float* uf = (float*)carve(NPTS * sizeof(float));
    float* vf = (float*)carve(NPTS * sizeof(float));
    _Float16* uh = (_Float16*)carve(NPTS * sizeof(_Float16));
    _Float16* vh = (_Float16*)carve(NPTS * sizeof(_Float16));
    float* partial = (float*)carve((size_t)JSEG * NPTS * sizeof(float));     // 256 KB
    (void)ws_size; (void)in_sizes; (void)n_in; (void)out_size;

    // prep
    cvt_rows<<<NPTS / 8, 256, 0, stream>>>(src, xh, x2);
    cvt_rows<<<NPTS / 8, 256, 0, stream>>>(tgt, yh, y2);
    softmax8192<<<1, 256, 0, stream>>>(sdens, mu);
    softmax8192<<<1, 256, 0, stream>>>(tdens, nu);
    init_uv<<<NPTS / 256, 256, 0, stream>>>(uf, uh, out);

    // K and K^T via WMMA
    build_K<<<dim3(64, 512), 256, 0, stream>>>(xh, yh, x2, y2, K, KT);

    // Sinkhorn iterations: v = nu/(K^T u); u = mu/(K v)  (all matvecs via WMMA)
    for (int it = 0; it < NITER; ++it) {
        pass_partial<<<dim3(64, JSEG), 256, 0, stream>>>(KT, uh, partial);
        pass_finish<<<NPTS / 256, 256, 0, stream>>>(partial, nu, vf, vh);
        pass_partial<<<dim3(64, JSEG), 256, 0, stream>>>(K, vh, partial);
        pass_finish<<<NPTS / 256, 256, 0, stream>>>(partial, mu, uf, uh);
    }

    // exact loss via WMMA recompute of distances
    loss_kernel<<<dim3(64, 512), 256, 0, stream>>>(xh, yh, x2, y2, uf, vf, out);
}
